// RPN_36034775613629
// MI455X (gfx1250) — matrix operand order repaired and checked by reading, not compiled, not run on hardware
//
#include <hip/hip_runtime.h>
#include <stdint.h>

typedef _Float16 half_t;
typedef __attribute__((ext_vector_type(16))) _Float16 v16h;
typedef __attribute__((ext_vector_type(8)))  float    v8f;

union Frag16 { v16h v; uint4 q[2]; _Float16 h[16]; };
union Acc8   { v8f  v; float f[8]; };

#define SCALE_CLAMP 4.135166556742356f
#define NMS_THRESH  0.7f
#define IMG_SZ      1024.0f
#define CAND_TOTAL  4768   /* 1000*4 + 768 (P6 has only 16*16*3=768 anchors) */
#define FPAD        8192

__device__ __forceinline__ unsigned long long score_key(float s, unsigned idx) {
  unsigned u = __float_as_uint(s);
  unsigned k32 = u ^ ((u >> 31) ? 0xFFFFFFFFu : 0x80000000u);  // ascending order key
  k32 = ~k32;                                                  // invert -> descending via ascending sort
  return ((unsigned long long)k32 << 32) | idx;
}

// ---------------------------------------------------------------------------
// Packing kernels
// ---------------------------------------------------------------------------

// NCHW fp32 -> NHWC f16
__global__ void pack_x_kernel(const float* __restrict__ in, half_t* __restrict__ out,
                              int Nimg, int H, int W) {
  const int C = 256;
  size_t total = (size_t)Nimg * H * W * C;
  for (size_t gid = (size_t)blockIdx.x * blockDim.x + threadIdx.x; gid < total;
       gid += (size_t)gridDim.x * blockDim.x) {
    int c = (int)(gid % C);
    size_t rest = gid / C;
    int x = (int)(rest % W); rest /= W;
    int y = (int)(rest % H);
    int n = (int)(rest / H);
    out[gid] = (half_t)in[(((size_t)n * C + c) * H + y) * W + x];
  }
}

// w_conv [cout][cin][3][3] fp32 -> WpackT [tap][cout][cin] f16
__global__ void pack_w_kernel(const float* __restrict__ w, half_t* __restrict__ out) {
  int total = 9 * 256 * 256;
  for (int gid = blockIdx.x * blockDim.x + threadIdx.x; gid < total;
       gid += gridDim.x * blockDim.x) {
    int cin  = gid % 256;
    int rest = gid / 256;
    int cout = rest % 256;
    int tap  = rest / 256;
    int ky = tap / 3, kx = tap % 3;
    out[gid] = (half_t)w[(((size_t)cout * 256 + cin) * 3 + ky) * 3 + kx];
  }
}

// fuse w_obj (3x256) + w_delta (12x256) -> Whead [16][256] f16 ; bias likewise
__global__ void pack_head_kernel(const float* __restrict__ w_obj, const float* __restrict__ b_obj,
                                 const float* __restrict__ w_delta, const float* __restrict__ b_delta,
                                 half_t* __restrict__ wh, float* __restrict__ bh) {
  int k = threadIdx.x;  // 256 threads
  for (int n = 0; n < 16; ++n) {
    float v = 0.f;
    if (n < 3)       v = w_obj[n * 256 + k];
    else if (n < 15) v = w_delta[(n - 3) * 256 + k];
    wh[n * 256 + k] = (half_t)v;
  }
  if (k < 16) {
    float v = 0.f;
    if (k < 3)       v = b_obj[k];
    else if (k < 15) v = b_delta[k - 3];
    bh[k] = v;
  }
}

// ---------------------------------------------------------------------------
// 3x3 conv 256->256, SAME padding, ReLU: implicit GEMM with WMMA
//   M = N*H*W, N = 256 (cout), K = 2304 = 9 taps x 8 chunks of 32
//   Block: 256 threads (8 waves), 64x128 output tile, double-buffered LDS.
//   Wave: 32x32 subtile -> 2 A-frags x 2 B-frags -> 4 WMMAs per K step.
// ---------------------------------------------------------------------------
__global__ __launch_bounds__(256) void conv3x3_wmma_kernel(
    const half_t* __restrict__ X,     // [N][H][W][256] f16
    const half_t* __restrict__ Wt,    // [tap][cout][cin] f16
    const float*  __restrict__ bias,  // [256]
    half_t*       __restrict__ Y,     // [N][H][W][256] f16 (post-ReLU activations)
    int Nimg, int H, int W) {
  const int M = Nimg * H * W;
  __shared__ _Float16 As[2][64][40];   // [buf][m][k]
  __shared__ _Float16 Bs[2][128][40];  // [buf][n][k] = W[cout][cin]

  const int tid  = threadIdx.x;
  const int lane = tid & 31;
  const int wv   = tid >> 5;
  const int g    = lane >> 4;      // half-wave group
  const int ln   = lane & 15;

  const int m0 = blockIdx.x * 64;
  const int n0 = blockIdx.y * 128;

  // A-tile loader coords: row = tid/4 (0..63), 8 halves per thread
  const int ami = tid >> 2, aseg = tid & 3;
  const int ap = m0 + ami;
  const bool aval = ap < M;
  int an = 0, ay = 0, ax = 0;
  if (aval) {
    an = ap / (H * W);
    int r = ap - an * H * W;
    ay = r / W;
    ax = r - ay * W;
  }
  // B-tile loader coords: row = tid/2 (0..127), 16 halves per thread
  const int bn = tid >> 1, bseg = tid & 1;

  const int msub = (wv & 1) * 32;
  const int nsub = (wv >> 1) * 32;

  Acc8 c00, c01, c10, c11;
#pragma unroll
  for (int i = 0; i < 8; ++i) { c00.f[i] = 0.f; c01.f[i] = 0.f; c10.f[i] = 0.f; c11.f[i] = 0.f; }

  auto gload = [&](int kk, uint4& av, uint4& bv0, uint4& bv1) {
    const int tap  = kk >> 3;
    const int cinb = (kk & 7) << 5;
    const int dy = tap / 3 - 1, dx = tap % 3 - 1;
    av = make_uint4(0u, 0u, 0u, 0u);
    if (aval) {
      int yy = ay + dy, xx = ax + dx;
      if (yy >= 0 && yy < H && xx >= 0 && xx < W) {
        av = *(const uint4*)(X + ((((size_t)an * H + yy) * W + xx) * 256 + cinb + aseg * 8));
      }
    }
    const half_t* wp = Wt + (((size_t)tap * 256 + n0 + bn) * 256 + cinb + bseg * 16);
    bv0 = *(const uint4*)(wp);
    bv1 = *(const uint4*)(wp + 8);
  };

  // prologue: stage K-step 0 into buffer 0
  {
    uint4 av, bv0, bv1;
    gload(0, av, bv0, bv1);
    *(uint4*)(&As[0][ami][aseg * 8])      = av;
    *(uint4*)(&Bs[0][bn][bseg * 16])      = bv0;
    *(uint4*)(&Bs[0][bn][bseg * 16 + 8])  = bv1;
  }
  __syncthreads();

  for (int kk = 0; kk < 72; ++kk) {
    const int cur = kk & 1, nxt = cur ^ 1;
    const bool more = (kk + 1) < 72;

    // issue next K-step's global loads (latency overlapped with WMMAs below)
    uint4 nav, nbv0, nbv1;
    if (more) gload(kk + 1, nav, nbv0, nbv1);

    // fragments per documented 16-bit A/B lane layouts
    Frag16 a0, a1, b0, b1;
    a0.q[0] = *(const uint4*)(&As[cur][msub + ln][g * 8]);        // K = g*8 + 0..7
    a0.q[1] = *(const uint4*)(&As[cur][msub + ln][16 + g * 8]);   // K = 16 + g*8 + 0..7
    a1.q[0] = *(const uint4*)(&As[cur][msub + 16 + ln][g * 8]);
    a1.q[1] = *(const uint4*)(&As[cur][msub + 16 + ln][16 + g * 8]);
    b0.q[0] = *(const uint4*)(&Bs[cur][nsub + ln][g * 8]);        // K = g*8 + 0..7
    b0.q[1] = *(const uint4*)(&Bs[cur][nsub + ln][16 + g * 8]);   // K = 16 + g*8 + 0..7
    b1.q[0] = *(const uint4*)(&Bs[cur][nsub + 16 + ln][g * 8]);
    b1.q[1] = *(const uint4*)(&Bs[cur][nsub + 16 + ln][16 + g * 8]);

    c00.v = __builtin_amdgcn_wmma_f32_16x16x32_f16(false, a0.v, false, b0.v, (short)0, c00.v, false, false);
    c01.v = __builtin_amdgcn_wmma_f32_16x16x32_f16(false, a0.v, false, b1.v, (short)0, c01.v, false, false);
    c10.v = __builtin_amdgcn_wmma_f32_16x16x32_f16(false, a1.v, false, b0.v, (short)0, c10.v, false, false);
    c11.v = __builtin_amdgcn_wmma_f32_16x16x32_f16(false, a1.v, false, b1.v, (short)0, c11.v, false, false);

    if (more) {
      *(uint4*)(&As[nxt][ami][aseg * 8])     = nav;
      *(uint4*)(&Bs[nxt][bn][bseg * 16])     = nbv0;
      *(uint4*)(&Bs[nxt][bn][bseg * 16 + 8]) = nbv1;
    }
    __syncthreads();
  }

  // epilogue: bias + ReLU, store f16 NHWC
  const int col0 = n0 + nsub + ln;
  const int col1 = col0 + 16;
  const float bz0 = bias[col0];
  const float bz1 = bias[col1];
#pragma unroll
  for (int r = 0; r < 8; ++r) {
    const int mr0 = m0 + msub + r + 8 * g;
    const int mr1 = mr0 + 16;
    if (mr0 < M) {
      float v0 = c00.f[r] + bz0; v0 = v0 > 0.f ? v0 : 0.f;
      float v1 = c01.f[r] + bz1; v1 = v1 > 0.f ? v1 : 0.f;
      Y[(size_t)mr0 * 256 + col0] = (half_t)v0;
      Y[(size_t)mr0 * 256 + col1] = (half_t)v1;
    }
    if (mr1 < M) {
      float v0 = c10.f[r] + bz0; v0 = v0 > 0.f ? v0 : 0.f;
      float v1 = c11.f[r] + bz1; v1 = v1 > 0.f ? v1 : 0.f;
      Y[(size_t)mr1 * 256 + col0] = (half_t)v0;
      Y[(size_t)mr1 * 256 + col1] = (half_t)v1;
    }
  }
}

// ---------------------------------------------------------------------------
// Head: [M,256] x [256,16] -> 16 outputs/pixel (3 logits + 12 deltas), then
// anchor generation + delta decode. One wave per 16-pixel tile, K=256 -> 8 WMMAs.
// ---------------------------------------------------------------------------
__global__ __launch_bounds__(256) void head_wmma_kernel(
    const half_t* __restrict__ Act,   // [M][256] f16
    const half_t* __restrict__ Wh,    // [16][256] f16
    const float*  __restrict__ Bh,    // [16]
    float* __restrict__ scores,       // [Nimg][HWA]
    float* __restrict__ boxes,        // [Nimg][HWA][4]
    int Nimg, int H, int W, float stride, float asize) {
  const int M = Nimg * H * W;
  __shared__ float tile[8][16][17];

  const int tid  = threadIdx.x;
  const int lane = tid & 31;
  const int wv   = tid >> 5;
  const int g    = lane >> 4;
  const int ln   = lane & 15;
  const int pbase = (blockIdx.x * 8 + wv) * 16;

  // B fragments: lane ln holds output column ln; K = g*8 + 0..7 and 16 + g*8 + 0..7
  Frag16 bf[8];
#pragma unroll
  for (int kb = 0; kb < 8; ++kb) {
    const half_t* src = Wh + ln * 256 + kb * 32 + g * 8;
    bf[kb].q[0] = *(const uint4*)(src);
    bf[kb].q[1] = *(const uint4*)(src + 16);
  }

  Acc8 c;
#pragma unroll
  for (int i = 0; i < 8; ++i) c.f[i] = 0.f;

  int p = pbase + ln;
  if (p >= M) p = M - 1;  // clamp; results discarded below
  const half_t* arow = Act + (size_t)p * 256;
#pragma unroll
  for (int kb = 0; kb < 8; ++kb) {
    Frag16 a;
    a.q[0] = *(const uint4*)(arow + kb * 32 + g * 8);
    a.q[1] = *(const uint4*)(arow + kb * 32 + 16 + g * 8);
    c.v = __builtin_amdgcn_wmma_f32_16x16x32_f16(false, a.v, false, bf[kb].v, (short)0, c.v, false, false);
  }

  const float bb = Bh[ln];
#pragma unroll
  for (int r = 0; r < 8; ++r) tile[wv][r + 8 * g][ln] = c.f[r] + bb;
  __syncthreads();

  if (tid < 128) {
    const int w2 = tid >> 4, m = tid & 15;
    const int p2 = (blockIdx.x * 8 + w2) * 16 + m;
    if (p2 < M) {
      float v[15];
#pragma unroll
      for (int j = 0; j < 15; ++j) v[j] = tile[w2][m][j];
      const int HW = H * W;
      const int n_img = p2 / HW;
      int rem = p2 - n_img * HW;
      const int y = rem / W;
      const int x = rem - y * W;
      const float cx = x * stride, cy = y * stride;
      const float ratios[3] = {0.5f, 1.0f, 2.0f};
      const int HWA = HW * 3;
      for (int a = 0; a < 3; ++a) {
        float wa = asize / sqrtf(ratios[a]);
        float ha = wa * ratios[a];
        float dxv = v[3 + a * 4 + 0];
        float dyv = v[3 + a * 4 + 1];
        float dwv = fminf(v[3 + a * 4 + 2], SCALE_CLAMP);
        float dhv = fminf(v[3 + a * 4 + 3], SCALE_CLAMP);
        float pcx = dxv * wa + cx;
        float pcy = dyv * ha + cy;
        float pw  = expf(dwv) * wa;
        float ph  = expf(dhv) * ha;
        int idx = (y * W + x) * 3 + a;
        scores[(size_t)n_img * HWA + idx] = v[a];
        float4 bx = make_float4(pcx - 0.5f * pw, pcy - 0.5f * ph,
                                pcx + 0.5f * pw, pcy + 0.5f * ph);
        *(float4*)(boxes + ((size_t)n_img * HWA + idx) * 4) = bx;
      }
    }
  }
}

// ---------------------------------------------------------------------------
// Top-k via bitonic sort on (flipped-score << 32 | index) keys
// ---------------------------------------------------------------------------
__global__ void build_keys_kernel(const float* __restrict__ scores,
                                  unsigned long long* __restrict__ keys,
                                  int HWA, int PAD) {
  int total = 2 * PAD;
  for (int gid = blockIdx.x * blockDim.x + threadIdx.x; gid < total;
       gid += gridDim.x * blockDim.x) {
    int img = gid / PAD, i = gid - img * PAD;
    unsigned long long key = 0xFFFFFFFFFFFFFFFFull;
    if (i < HWA) key = score_key(scores[(size_t)img * HWA + i], (unsigned)i);
    keys[(size_t)img * PAD + i] = key;
  }
}

__global__ void bitonic_step_kernel(unsigned long long* __restrict__ d,
                                    int n, int nseg, int k, int j) {
  int gid = blockIdx.x * blockDim.x + threadIdx.x;
  int total = nseg * n;
  if (gid >= total) return;
  int seg = gid / n, i = gid - seg * n;
  int l = i ^ j;
  if (l <= i) return;
  unsigned long long* base = d + (size_t)seg * n;
  unsigned long long a = base[i], b = base[l];
  bool up = ((i & k) == 0);
  if (up ? (a > b) : (a < b)) { base[i] = b; base[l] = a; }
}

// ---------------------------------------------------------------------------
// Gather per-level top-k into candidate arrays (with clip, valid, level offset)
// ---------------------------------------------------------------------------
struct LvlArg {
  const unsigned long long* keys;
  const float* scores;
  const float* boxes;
  int HWA;
  int PAD;
  int kTake;   // min(1000, HWA)
  int start;   // running offset into candidate list
};

__global__ void gather_cand_kernel(LvlArg L0, LvlArg L1, LvlArg L2, LvlArg L3, LvlArg L4,
                                   float* __restrict__ cbox, float* __restrict__ coff,
                                   float* __restrict__ cscore, float* __restrict__ cvalid,
                                   unsigned long long* __restrict__ fkeys) {
  LvlArg Ls[5] = {L0, L1, L2, L3, L4};
  int total = 2 * FPAD;
  for (int gid = blockIdx.x * blockDim.x + threadIdx.x; gid < total;
       gid += gridDim.x * blockDim.x) {
    int img = gid / FPAD, t = gid - img * FPAD;
    if (t < CAND_TOTAL) {
      int lvl = 0;
      while (lvl < 4 && t >= Ls[lvl + 1].start) ++lvl;
      LvlArg L = Ls[lvl];
      int r = t - L.start;
      unsigned long long key = L.keys[(size_t)img * L.PAD + r];
      int idx = (int)(key & 0xFFFFFFFFu);
      if (idx >= L.HWA) idx = 0;
      float s = L.scores[(size_t)img * L.HWA + idx];
      const float* bp = L.boxes + ((size_t)img * L.HWA + idx) * 4;
      float x0 = fminf(fmaxf(bp[0], 0.f), IMG_SZ);
      float y0 = fminf(fmaxf(bp[1], 0.f), IMG_SZ);
      float x1 = fminf(fmaxf(bp[2], 0.f), IMG_SZ);
      float y1 = fminf(fmaxf(bp[3], 0.f), IMG_SZ);
      float valid = ((x1 - x0) > 0.f && (y1 - y0) > 0.f) ? 1.f : 0.f;
      float off = (float)lvl * (IMG_SZ + 1.0f);
      size_t base = ((size_t)img * CAND_TOTAL + t) * 4;
      cbox[base + 0] = x0; cbox[base + 1] = y0; cbox[base + 2] = x1; cbox[base + 3] = y1;
      coff[base + 0] = x0 + off; coff[base + 1] = y0 + off;
      coff[base + 2] = x1 + off; coff[base + 3] = y1 + off;
      cscore[(size_t)img * CAND_TOTAL + t] = s;
      cvalid[(size_t)img * CAND_TOTAL + t] = valid;
      fkeys[(size_t)img * FPAD + t] = score_key(s, (unsigned)t);
    } else {
      fkeys[(size_t)img * FPAD + t] = 0xFFFFFFFFFFFFFFFFull;
    }
  }
}

__global__ void gather_sorted_kernel(const unsigned long long* __restrict__ fkeys,
                                     const float* __restrict__ cbox, const float* __restrict__ coff,
                                     const float* __restrict__ cscore, const float* __restrict__ cvalid,
                                     float* __restrict__ sbox, float* __restrict__ soff,
                                     float* __restrict__ sscore, float* __restrict__ svalid) {
  int total = 2 * CAND_TOTAL;
  for (int gid = blockIdx.x * blockDim.x + threadIdx.x; gid < total;
       gid += gridDim.x * blockDim.x) {
    int img = gid / CAND_TOTAL, t = gid - img * CAND_TOTAL;
    unsigned long long key = fkeys[(size_t)img * FPAD + t];
    int idx = (int)(key & 0xFFFFFFFFu);
    if (idx >= CAND_TOTAL) idx = 0;
    size_t src = ((size_t)img * CAND_TOTAL + idx) * 4;
    size_t dst = ((size_t)img * CAND_TOTAL + t) * 4;
#pragma unroll
    for (int i = 0; i < 4; ++i) { sbox[dst + i] = cbox[src + i]; soff[dst + i] = coff[src + i]; }
    sscore[(size_t)img * CAND_TOTAL + t] = cscore[(size_t)img * CAND_TOTAL + idx];
    svalid[(size_t)img * CAND_TOTAL + t] = cvalid[(size_t)img * CAND_TOTAL + idx];
  }
}

// ---------------------------------------------------------------------------
// Greedy NMS (one block per image) + emission of top-1000 (boxes, scores)
// ---------------------------------------------------------------------------
__global__ __launch_bounds__(256) void nms_emit_kernel(
    const float* __restrict__ sbox, const float* __restrict__ soff,
    const float* __restrict__ sscore, const float* __restrict__ svalid,
    float* __restrict__ outBoxes, float* __restrict__ outScores) {
  __shared__ unsigned char keep[CAND_TOTAL];
  __shared__ float cur[4];
  const int img = blockIdx.x;
  const int tid = threadIdx.x;
  const float* OB = soff + (size_t)img * CAND_TOTAL * 4;

  for (int t = tid; t < CAND_TOTAL; t += blockDim.x) keep[t] = 1;
  __syncthreads();

  for (int i = 0; i < CAND_TOTAL; ++i) {
    if (tid == 0) {
#pragma unroll
      for (int c = 0; c < 4; ++c) cur[c] = OB[(size_t)i * 4 + c];
    }
    __syncthreads();
    if (keep[i]) {
      const float ax0 = cur[0], ay0 = cur[1], ax1 = cur[2], ay1 = cur[3];
      const float areaA = (ax1 - ax0) * (ay1 - ay0);
      for (int jj = i + 1 + tid; jj < CAND_TOTAL; jj += blockDim.x) {
        if (!keep[jj]) continue;
        float bx0 = OB[(size_t)jj * 4 + 0], by0 = OB[(size_t)jj * 4 + 1];
        float bx1 = OB[(size_t)jj * 4 + 2], by1 = OB[(size_t)jj * 4 + 3];
        float ix0 = fmaxf(ax0, bx0), iy0 = fmaxf(ay0, by0);
        float ix1 = fminf(ax1, bx1), iy1 = fminf(ay1, by1);
        float iw = fmaxf(ix1 - ix0, 0.f), ih = fmaxf(iy1 - iy0, 0.f);
        float inter = iw * ih;
        float areaB = (bx1 - bx0) * (by1 - by0);
        float uni = areaA + areaB - inter;
        float iou = uni > 0.f ? inter / uni : 0.f;
        if (iou > NMS_THRESH) keep[jj] = 0;
      }
    }
    __syncthreads();
  }

  if (tid == 0) {
    int cnt = 0;
    for (int t = 0; t < CAND_TOTAL && cnt < 1000; ++t) {
      if (keep[t] && svalid[(size_t)img * CAND_TOTAL + t] > 0.5f) {
        const float* bp = sbox + ((size_t)img * CAND_TOTAL + t) * 4;
        size_t o = ((size_t)img * 1000 + cnt) * 4;
#pragma unroll
        for (int c = 0; c < 4; ++c) outBoxes[o + c] = bp[c];
        outScores[(size_t)img * 1000 + cnt] = sscore[(size_t)img * CAND_TOTAL + t];
        ++cnt;
      }
    }
    for (; cnt < 1000; ++cnt) {
      size_t o = ((size_t)img * 1000 + cnt) * 4;
#pragma unroll
      for (int c = 0; c < 4; ++c) outBoxes[o + c] = 0.f;
      outScores[(size_t)img * 1000 + cnt] = -__builtin_inff();
    }
  }
}

// ---------------------------------------------------------------------------
// Host driver
// ---------------------------------------------------------------------------
extern "C" void kernel_launch(void* const* d_in, const int* in_sizes, int n_in,
                              void* d_out, int out_size, void* d_ws, size_t ws_size,
                              hipStream_t stream) {
  (void)in_sizes; (void)n_in; (void)out_size; (void)ws_size;
  const float* feat[5] = {(const float*)d_in[0], (const float*)d_in[1], (const float*)d_in[2],
                          (const float*)d_in[3], (const float*)d_in[4]};
  const float* w_conv  = (const float*)d_in[5];
  const float* b_conv  = (const float*)d_in[6];
  const float* w_obj   = (const float*)d_in[7];
  const float* b_obj   = (const float*)d_in[8];
  const float* w_delta = (const float*)d_in[9];
  const float* b_delta = (const float*)d_in[10];

  static const int   Hs[5]  = {256, 128, 64, 32, 16};
  static const int   Wsz[5] = {256, 128, 64, 32, 16};
  static const float STR[5] = {4.f, 8.f, 16.f, 32.f, 64.f};
  static const float ASZ[5] = {32.f, 64.f, 128.f, 256.f, 512.f};
  static const int   PADv[5] = {262144, 65536, 16384, 4096, 1024};
  static const int   KT[5]  = {1000, 1000, 1000, 1000, 768};
  static const int   ST[5]  = {0, 1000, 2000, 3000, 4000};

  char* ws = (char*)d_ws;
  size_t off = 0;
  auto alloc = [&](size_t bytes) -> void* {
    void* p = ws + off;
    off = (off + bytes + 255) & ~(size_t)255;
    return p;
  };

  half_t *xf[5], *act[5];
  float *sc[5], *bx[5];
  unsigned long long* keys[5];
  int HWA[5];
  for (int l = 0; l < 5; ++l) {
    const int H = Hs[l], W = Wsz[l];
    HWA[l] = H * W * 3;
    size_t nh = (size_t)2 * H * W * 256;
    xf[l]   = (half_t*)alloc(nh * sizeof(half_t));
    act[l]  = (half_t*)alloc(nh * sizeof(half_t));
    sc[l]   = (float*)alloc((size_t)2 * HWA[l] * sizeof(float));
    bx[l]   = (float*)alloc((size_t)2 * HWA[l] * 4 * sizeof(float));
    keys[l] = (unsigned long long*)alloc((size_t)2 * PADv[l] * sizeof(unsigned long long));
  }
  half_t* wpk = (half_t*)alloc((size_t)9 * 256 * 256 * sizeof(half_t));
  half_t* wh  = (half_t*)alloc((size_t)16 * 256 * sizeof(half_t));
  float*  bh  = (float*)alloc(16 * sizeof(float));
  float* cbox   = (float*)alloc((size_t)2 * CAND_TOTAL * 4 * sizeof(float));
  float* coff   = (float*)alloc((size_t)2 * CAND_TOTAL * 4 * sizeof(float));
  float* cscore = (float*)alloc((size_t)2 * CAND_TOTAL * sizeof(float));
  float* cvalid = (float*)alloc((size_t)2 * CAND_TOTAL * sizeof(float));
  unsigned long long* fkeys = (unsigned long long*)alloc((size_t)2 * FPAD * sizeof(unsigned long long));
  float* sbox   = (float*)alloc((size_t)2 * CAND_TOTAL * 4 * sizeof(float));
  float* soff   = (float*)alloc((size_t)2 * CAND_TOTAL * 4 * sizeof(float));
  float* sscore = (float*)alloc((size_t)2 * CAND_TOTAL * sizeof(float));
  float* svalid = (float*)alloc((size_t)2 * CAND_TOTAL * sizeof(float));

  // weight packing
  pack_w_kernel<<<(9 * 256 * 256 + 255) / 256, 256, 0, stream>>>(w_conv, wpk);
  pack_head_kernel<<<1, 256, 0, stream>>>(w_obj, b_obj, w_delta, b_delta, wh, bh);

  for (int l = 0; l < 5; ++l) {
    const int H = Hs[l], W = Wsz[l];
    const int M = 2 * H * W;
    size_t totalX = (size_t)M * 256;
    int gx = (int)((totalX + 255) / 256);
    if (gx > 8192) gx = 8192;
    pack_x_kernel<<<gx, 256, 0, stream>>>(feat[l], xf[l], 2, H, W);

    dim3 cgrid((M + 63) / 64, 2);  // 64 rows x 128 couts per block
    conv3x3_wmma_kernel<<<cgrid, 256, 0, stream>>>(xf[l], wpk, b_conv, act[l], 2, H, W);

    head_wmma_kernel<<<(M + 127) / 128, 256, 0, stream>>>(act[l], wh, bh, sc[l], bx[l],
                                                          2, H, W, STR[l], ASZ[l]);

    int PAD = PADv[l];
    build_keys_kernel<<<(2 * PAD + 255) / 256, 256, 0, stream>>>(sc[l], keys[l], HWA[l], PAD);
    for (int k = 2; k <= PAD; k <<= 1)
      for (int j = k >> 1; j > 0; j >>= 1)
        bitonic_step_kernel<<<(2 * PAD + 255) / 256, 256, 0, stream>>>(keys[l], PAD, 2, k, j);
  }

  LvlArg La[5];
  for (int l = 0; l < 5; ++l)
    La[l] = LvlArg{keys[l], sc[l], bx[l], HWA[l], PADv[l], KT[l], ST[l]};
  gather_cand_kernel<<<(2 * FPAD + 255) / 256, 256, 0, stream>>>(
      La[0], La[1], La[2], La[3], La[4], cbox, coff, cscore, cvalid, fkeys);

  for (int k = 2; k <= FPAD; k <<= 1)
    for (int j = k >> 1; j > 0; j >>= 1)
      bitonic_step_kernel<<<(2 * FPAD + 255) / 256, 256, 0, stream>>>(fkeys, FPAD, 2, k, j);

  gather_sorted_kernel<<<(2 * CAND_TOTAL + 255) / 256, 256, 0, stream>>>(
      fkeys, cbox, coff, cscore, cvalid, sbox, soff, sscore, svalid);

  float* outBoxes  = (float*)d_out;           // [2][1000][4]
  float* outScores = (float*)d_out + 8000;    // [2][1000]
  nms_emit_kernel<<<2, 256, 0, stream>>>(sbox, soff, sscore, svalid, outBoxes, outScores);
}
